// MambaBlockPure_55722905699102
// MI455X (gfx1250) — compile-verified, matrix operand order checked by reading, not run on hardware
//
#include <hip/hip_runtime.h>
#include <hip/hip_bf16.h>
#include <math.h>

// ---------------------------------------------------------------------------
// Mamba block forward for MI455X (gfx1250, wave32, WMMA).
// GEMMs run in bf16 via v_wmma_f32_16x16x32_bf16 with pre-transposed,
// zero-padded bf16 weights ([Npad][K]); tiles are staged to LDS with the
// CDNA5 async global->LDS path (ASYNCcnt / s_wait_asynccnt) and double
// buffering so DMA overlaps the 16-WMMA compute per K-step.
// The recurrent selective scan is a register-resident per-channel recurrence.
// ---------------------------------------------------------------------------

typedef __bf16 bf16;
typedef __attribute__((ext_vector_type(16))) __bf16 v16bf;
typedef __attribute__((ext_vector_type(8)))  __bf16 bf16x8;
typedef __attribute__((ext_vector_type(8)))  float  v8f;
typedef __attribute__((__vector_size__(16))) int    vi4;   // int4 vector

#if __has_builtin(__builtin_amdgcn_global_load_async_to_lds_b128)
#define USE_ASYNC_LDS 1
#else
#define USE_ASYNC_LDS 0
#endif

#if USE_ASYNC_LDS
// Async copy: param0 = global (AS1) int4*, param1 = LDS (AS3) int4*.
#define ASYNC_CP_B128(gp, lp)                                             \
  __builtin_amdgcn_global_load_async_to_lds_b128(                         \
      (__attribute__((address_space(1))) vi4*)(void*)(gp),                \
      (__attribute__((address_space(3))) vi4*)(void*)(lp), 0, 0)
// Wait for all but the most-recently-issued group (6 chunks) / everything.
#define ASYNC_WAIT_GROUP() asm volatile("s_wait_asynccnt 0x6" ::: "memory")
#define ASYNC_WAIT_ALL()   asm volatile("s_wait_asynccnt 0x0" ::: "memory")
#else
#define ASYNC_CP_B128(gp, lp) (*(uint4*)(lp) = *(const uint4*)(gp))
#define ASYNC_WAIT_GROUP()
#define ASYNC_WAIT_ALL()
#endif

constexpr int D_MODEL  = 1024;
constexpr int D_STATE  = 16;
constexpr int D_CONV   = 4;
constexpr int D_INNER  = 2048;
constexpr int DT_RANK  = 64;
constexpr int BATCH    = 4;
constexpr int SEQ_LEN  = 2048;
constexpr int M_TOK    = BATCH * SEQ_LEN;          // 8192 rows in every GEMM
constexpr int XDBL_N   = DT_RANK + 2 * D_STATE;    // 96
constexpr int XDBL_NP  = 128;                      // padded to N-tile

// ----------------------------- elementwise ---------------------------------

__global__ void cast_f32_to_bf16(const float* __restrict__ src,
                                 bf16* __restrict__ dst, int n) {
  int i = blockIdx.x * blockDim.x + threadIdx.x;
  if (i < n) dst[i] = (bf16)src[i];
}

// Weights: src f32 [K][N] row-major  ->  dst bf16 [Npad][K] (transposed,
// zero-padded rows N..Npad). One-time pass; makes GEMM B-tile loads b128.
__global__ void cast_transpose_pad_bf16(const float* __restrict__ src,
                                        bf16* __restrict__ dst,
                                        int K, int N, int Npad) {
  int i = blockIdx.x * blockDim.x + threadIdx.x;
  if (i >= Npad * K) return;
  int n = i / K, k = i % K;
  dst[i] = (n < N) ? (bf16)src[(size_t)k * N + n] : (bf16)0.0f;
}

// dst[r][c] = (bf16) src[r][col0 + c]   (strided column-slice cast)
__global__ void cast_slice_bf16(const float* __restrict__ src, int ld, int col0,
                                int ncols, bf16* __restrict__ dst, int rows) {
  int i = blockIdx.x * blockDim.x + threadIdx.x;
  if (i >= rows * ncols) return;
  int r = i / ncols, c = i % ncols;
  dst[i] = (bf16)src[(size_t)r * ld + col0 + c];
}

// causal depthwise conv1d + bias + SiLU, fused bf16 cast for the x_proj GEMM
__global__ void conv_silu_kernel(const float* __restrict__ xz,
                                 const float* __restrict__ conv_w,
                                 const float* __restrict__ conv_b,
                                 float* __restrict__ xconv,
                                 bf16* __restrict__ xconv_bf) {
  int i = blockIdx.x * blockDim.x + threadIdx.x;   // over M_TOK * D_INNER
  if (i >= M_TOK * D_INNER) return;
  int d   = i % D_INNER;
  int tok = i / D_INNER;
  int l   = tok % SEQ_LEN;
  int b   = tok / SEQ_LEN;
  float acc = conv_b[d];
#pragma unroll
  for (int k = 0; k < D_CONV; ++k) {
    int ls = l - (D_CONV - 1) + k;
    if (ls >= 0)
      acc += xz[((size_t)b * SEQ_LEN + ls) * (2 * D_INNER) + d] * conv_w[d * D_CONV + k];
  }
  float s = acc / (1.0f + __expf(-acc));           // SiLU
  xconv[i]    = s;
  xconv_bf[i] = (bf16)s;
}

// dt = softplus(dt_lin + bias), in place on the GEMM output
__global__ void dt_softplus_kernel(float* __restrict__ dt,
                                   const float* __restrict__ bias) {
  int i = blockIdx.x * blockDim.x + threadIdx.x;
  if (i >= M_TOK * D_INNER) return;
  float v = dt[i] + bias[i % D_INNER];
  dt[i] = (v > 20.0f) ? v : log1pf(__expf(v));
}

// y_gated = y * silu(z)  -> bf16 for the out_proj GEMM
__global__ void gate_cast_kernel(const float* __restrict__ y,
                                 const float* __restrict__ xz,
                                 bf16* __restrict__ yg) {
  int i = blockIdx.x * blockDim.x + threadIdx.x;
  if (i >= M_TOK * D_INNER) return;
  int d = i % D_INNER;
  size_t tok = (size_t)(i / D_INNER);
  float z = xz[tok * (2 * D_INNER) + D_INNER + d];
  float g = z / (1.0f + __expf(-z));
  yg[i] = (bf16)(y[i] * g);
}

// ----------------------------- WMMA GEMM -----------------------------------
// C[M,N] (f32) = A[M,K] (bf16 row-major) * Bt[Npad][K] (bf16, pre-transposed)
// Workgroup tile 256x128, 8 waves in 4x2; each wave owns 64x64 = 4x4 grid of
// 16x16 v_wmma_f32_16x16x32_bf16 accumulators. K stepped by 32, with
// double-buffered LDS tiles filled by async global->LDS DMA.
// Requires M % 256 == 0, K % 32 == 0, Npad % 128 == 0 (B zero-padded).

#define TILE_M 256
#define TILE_N 128
#define TILE_K 32

__launch_bounds__(256, 1)
__global__ void gemm_bf16_wmma(const bf16* __restrict__ A,
                               const bf16* __restrict__ Bt,
                               float* __restrict__ C, int M, int N, int K) {
  __shared__ __align__(16) bf16 As[2][TILE_M * TILE_K];    // 2 x 16 KB
  __shared__ __align__(16) bf16 Bs[2][TILE_N * TILE_K];    // 2 x  8 KB
  const int t    = threadIdx.x;
  const int lane = t & 31;
  const int wave = t >> 5;          // 0..7
  const int wm   = wave & 3;        // wave M offset: wm*64
  const int wn   = wave >> 2;       // wave N offset: wn*64
  const int m0   = blockIdx.x * TILE_M;
  const int n0   = blockIdx.y * TILE_N;

  // ISA 16-bit fragment layout (CDNA5 ISA 7.12.2): lane%16 = matrix row
  // (A) / col (B); lane/16 selects which 8-wide K-halves the lane holds.
  const int half = lane >> 4;
  const int r16  = lane & 15;

  // Per-thread 16B chunk coordinates (fixed across K-steps).
  const int crow = t >> 2;          // chunk row within tile
  const int ccol = (t & 3) << 3;    // chunk col (8 bf16 = 16B)

  // Issue one K-step's A+B tiles (6 x 16B per thread) into LDS buffer `buf`.
  auto issue_tiles = [&](int kidx, int buf) {
    const int k0 = kidx * TILE_K;
#pragma unroll
    for (int i = 0; i < 4; ++i) {               // A: 256x32, 4 chunks/thread
      const int row = crow + 64 * i;
      const bf16* gp = A + (size_t)(m0 + row) * K + k0 + ccol;
      bf16* lp = &As[buf][row * TILE_K + ccol];
      ASYNC_CP_B128(gp, lp);
    }
#pragma unroll
    for (int i = 0; i < 2; ++i) {               // B: 128x32, 2 chunks/thread
      const int row = crow + 64 * i;
      const bf16* gp = Bt + (size_t)(n0 + row) * K + k0 + ccol;
      bf16* lp = &Bs[buf][row * TILE_K + ccol];
      ASYNC_CP_B128(gp, lp);
    }
  };

  v8f acc[4][4] = {};
  const int nk = K / TILE_K;

  issue_tiles(0, 0);
  for (int i = 0; i < nk; ++i) {
    const int cur = i & 1;
    const bool more = (i + 1 < nk);
    if (more) issue_tiles(i + 1, cur ^ 1);   // DMA next tile during compute
    if (more) ASYNC_WAIT_GROUP(); else ASYNC_WAIT_ALL();
    __syncthreads();                         // tile i visible to all waves

    // Fragments: two ds_read_b128 each; 16 WMMA per K-step per wave.
    v16bf a[4];
#pragma unroll
    for (int tm = 0; tm < 4; ++tm) {
      const bf16* base = &As[cur][(wm * 64 + tm * 16 + r16) * TILE_K + half * 8];
      union { v16bf v; bf16x8 h[2]; } u;
      u.h[0] = *(const bf16x8*)(base);       // K in [half*8, half*8+8)
      u.h[1] = *(const bf16x8*)(base + 16);  // K in [16+half*8, ...)
      a[tm] = u.v;
    }
#pragma unroll
    for (int tn = 0; tn < 4; ++tn) {
      const bf16* base = &Bs[cur][(wn * 64 + tn * 16 + r16) * TILE_K + half * 8];
      union { v16bf v; bf16x8 h[2]; } u;
      u.h[0] = *(const bf16x8*)(base);
      u.h[1] = *(const bf16x8*)(base + 16);
      v16bf b = u.v;
#pragma unroll
      for (int tm = 0; tm < 4; ++tm)
        acc[tm][tn] = __builtin_amdgcn_wmma_f32_16x16x32_bf16(
            false, a[tm], false, b, (short)0, acc[tm][tn], false, false);
    }
    __syncthreads();   // all waves done reading buf `cur` before it's refilled
  }

  // Epilogue: C layout — lane%16 = column, VGPR r holds row r + 8*(lane/16).
#pragma unroll
  for (int tm = 0; tm < 4; ++tm)
#pragma unroll
    for (int tn = 0; tn < 4; ++tn) {
      int col = n0 + wn * 64 + tn * 16 + r16;
      if (col < N) {
#pragma unroll
        for (int r = 0; r < 8; ++r) {
          int row = m0 + wm * 64 + tm * 16 + r + 8 * half;
          C[(size_t)row * N + col] = acc[tm][tn][r];
        }
      }
    }
}

// ----------------------------- selective scan -------------------------------
// One thread per (batch, channel); 16 states in registers; B_t/C_t staged in
// LDS once per timestep (shared across the 256 channels of the block).

__launch_bounds__(256)
__global__ void selective_scan_kernel(const float* __restrict__ xconv,
                                      const float* __restrict__ dt,
                                      const float* __restrict__ xdbl,
                                      const float* __restrict__ A_log,
                                      const float* __restrict__ Dvec,
                                      float* __restrict__ y) {
  __shared__ float Bsh[D_STATE];
  __shared__ float Csh[D_STATE];
  const int b = blockIdx.y;
  const int d = blockIdx.x * blockDim.x + threadIdx.x;   // 0..D_INNER-1

  float a[D_STATE], h[D_STATE];
#pragma unroll
  for (int n = 0; n < D_STATE; ++n) {
    a[n] = -__expf(A_log[d * D_STATE + n]);
    h[n] = 0.0f;
  }
  const float Dd = Dvec[d];

  for (int l = 0; l < SEQ_LEN; ++l) {
    const size_t tok = (size_t)b * SEQ_LEN + l;
    if (threadIdx.x < 2 * D_STATE) {
      float v = xdbl[tok * XDBL_N + DT_RANK + threadIdx.x];
      if (threadIdx.x < D_STATE) Bsh[threadIdx.x] = v;
      else                       Csh[threadIdx.x - D_STATE] = v;
    }
    __syncthreads();

    const float xt  = xconv[tok * D_INNER + d];
    const float dtt = dt[tok * D_INNER + d];
    const float dtx = dtt * xt;
    float yt = 0.0f;
#pragma unroll
    for (int n = 0; n < D_STATE; ++n) {
      h[n] = __expf(a[n] * dtt) * h[n] + dtx * Bsh[n];
      yt  += h[n] * Csh[n];
    }
    y[tok * D_INNER + d] = yt + Dd * xt;
    __syncthreads();
  }
}

// ----------------------------- launch --------------------------------------

static inline size_t ws_take(size_t& off, size_t bytes) {
  size_t p = off;
  off = (off + bytes + 255) & ~(size_t)255;
  return p;
}

extern "C" void kernel_launch(void* const* d_in, const int* in_sizes, int n_in,
                              void* d_out, int out_size, void* d_ws, size_t ws_size,
                              hipStream_t stream) {
  const float* hs        = (const float*)d_in[0];  // (B,L,D_MODEL)
  const float* in_proj_w = (const float*)d_in[1];  // (D_MODEL, 2*D_INNER)
  const float* conv_w    = (const float*)d_in[2];  // (D_INNER, D_CONV)
  const float* conv_b    = (const float*)d_in[3];  // (D_INNER,)
  const float* x_proj_w  = (const float*)d_in[4];  // (D_INNER, 96)
  const float* dt_proj_w = (const float*)d_in[5];  // (DT_RANK, D_INNER)
  const float* dt_proj_b = (const float*)d_in[6];  // (D_INNER,)
  const float* A_log     = (const float*)d_in[7];  // (D_INNER, D_STATE)
  const float* Dvec      = (const float*)d_in[8];  // (D_INNER,)
  const float* out_proj_w= (const float*)d_in[9];  // (D_INNER, D_MODEL)
  float* out = (float*)d_out;                      // (B,L,D_MODEL) f32

  char* ws = (char*)d_ws;
  size_t off = 0;
  bf16*  hs_bf     = (bf16*)(ws + ws_take(off, (size_t)M_TOK * D_MODEL * 2));
  bf16*  inwT      = (bf16*)(ws + ws_take(off, (size_t)(2 * D_INNER) * D_MODEL * 2));
  bf16*  xprojwT   = (bf16*)(ws + ws_take(off, (size_t)XDBL_NP * D_INNER * 2));
  bf16*  dtprojwT  = (bf16*)(ws + ws_take(off, (size_t)D_INNER * DT_RANK * 2));
  bf16*  outwT     = (bf16*)(ws + ws_take(off, (size_t)D_MODEL * D_INNER * 2));
  float* xz        = (float*)(ws + ws_take(off, (size_t)M_TOK * 2 * D_INNER * 4));
  float* xconv     = (float*)(ws + ws_take(off, (size_t)M_TOK * D_INNER * 4));
  bf16*  xconv_bf  = (bf16*)(ws + ws_take(off, (size_t)M_TOK * D_INNER * 2));
  float* xdbl      = (float*)(ws + ws_take(off, (size_t)M_TOK * XDBL_N * 4));
  bf16*  dtraw_bf  = (bf16*)(ws + ws_take(off, (size_t)M_TOK * DT_RANK * 2));
  float* dtbuf     = (float*)(ws + ws_take(off, (size_t)M_TOK * D_INNER * 4));
  float* ybuf      = (float*)(ws + ws_take(off, (size_t)M_TOK * D_INNER * 4));
  bf16*  ygbf      = (bf16*)(ws + ws_take(off, (size_t)M_TOK * D_INNER * 2));
  (void)ws_size; (void)n_in; (void)in_sizes; (void)out_size;

  const int TB = 256;
  auto blocks = [&](size_t n) { return (unsigned)((n + TB - 1) / TB); };

  // 1) bf16 conversions (activations plain; weights transposed + padded)
  cast_f32_to_bf16<<<blocks((size_t)M_TOK * D_MODEL), TB, 0, stream>>>(
      hs, hs_bf, M_TOK * D_MODEL);
  cast_transpose_pad_bf16<<<blocks((size_t)(2 * D_INNER) * D_MODEL), TB, 0, stream>>>(
      in_proj_w, inwT, D_MODEL, 2 * D_INNER, 2 * D_INNER);
  cast_transpose_pad_bf16<<<blocks((size_t)XDBL_NP * D_INNER), TB, 0, stream>>>(
      x_proj_w, xprojwT, D_INNER, XDBL_N, XDBL_NP);
  cast_transpose_pad_bf16<<<blocks((size_t)D_INNER * DT_RANK), TB, 0, stream>>>(
      dt_proj_w, dtprojwT, DT_RANK, D_INNER, D_INNER);
  cast_transpose_pad_bf16<<<blocks((size_t)D_MODEL * D_INNER), TB, 0, stream>>>(
      out_proj_w, outwT, D_INNER, D_MODEL, D_MODEL);

  // 2) xz = hs @ in_proj_w   [8192 x 4096, K=1024]
  gemm_bf16_wmma<<<dim3(M_TOK / TILE_M, (2 * D_INNER) / TILE_N), TB, 0, stream>>>(
      hs_bf, inwT, xz, M_TOK, 2 * D_INNER, D_MODEL);

  // 3) causal conv + SiLU on x half
  conv_silu_kernel<<<blocks((size_t)M_TOK * D_INNER), TB, 0, stream>>>(
      xz, conv_w, conv_b, xconv, xconv_bf);

  // 4) x_dbl = xconv @ x_proj_w   [8192 x 96 (pad 128), K=2048]
  gemm_bf16_wmma<<<dim3(M_TOK / TILE_M, XDBL_NP / TILE_N), TB, 0, stream>>>(
      xconv_bf, xprojwT, xdbl, M_TOK, XDBL_N, D_INNER);

  // 5) dt_raw slice -> bf16; dt_lin = dt_raw @ dt_proj_w  [8192 x 2048, K=64]
  cast_slice_bf16<<<blocks((size_t)M_TOK * DT_RANK), TB, 0, stream>>>(
      xdbl, XDBL_N, 0, DT_RANK, dtraw_bf, M_TOK);
  gemm_bf16_wmma<<<dim3(M_TOK / TILE_M, D_INNER / TILE_N), TB, 0, stream>>>(
      dtraw_bf, dtprojwT, dtbuf, M_TOK, D_INNER, DT_RANK);
  dt_softplus_kernel<<<blocks((size_t)M_TOK * D_INNER), TB, 0, stream>>>(dtbuf, dt_proj_b);

  // 6) selective scan (sequential over L, parallel over B x D_INNER)
  selective_scan_kernel<<<dim3(D_INNER / TB, BATCH), TB, 0, stream>>>(
      xconv, dtbuf, xdbl, A_log, Dvec, ybuf);

  // 7) gate with silu(z), cast to bf16
  gate_cast_kernel<<<blocks((size_t)M_TOK * D_INNER), TB, 0, stream>>>(ybuf, xz, ygbf);

  // 8) out = y_gated @ out_proj_w  [8192 x 1024, K=2048] -> d_out (f32)
  gemm_bf16_wmma<<<dim3(M_TOK / TILE_M, D_MODEL / TILE_N), TB, 0, stream>>>(
      ygbf, outwT, out, M_TOK, D_MODEL, D_INNER);
}
